// GRUEncoder_30812095381521
// MI455X (gfx1250) — compile-verified, matrix operand order checked by reading, not run on hardware
//
#include <hip/hip_runtime.h>
#include <hip/hip_bf16.h>

// ---- problem constants (match reference) ----
#define S_  128
#define B_  64
#define V_  32000
#define EH_ 512
#define EO_ 512
#define H_  1024
#define O_  1024
#define G3H_ (3 * H_)
#define SB_ (S_ * B_)        // 8192 token rows

typedef __attribute__((ext_vector_type(16))) __bf16       v16bf;
typedef __attribute__((ext_vector_type(8)))  float        v8f;
typedef __attribute__((ext_vector_type(4)))  unsigned int v4u;

// fp32 -> bf16 round-to-nearest-even
__device__ __forceinline__ unsigned short f2bf(float f) {
    unsigned int u = __float_as_uint(f);
    unsigned int r = 0x7FFFu + ((u >> 16) & 1u);
    return (unsigned short)((u + r) >> 16);
}

// Load one 16-bf16 WMMA fragment from a K-contiguous row (A or B operand).
// Per CDNA5 ISA 16-bit 16x32 operand layout:
//   half==0: elems 0..7 -> K=k0+0..7,  elems 8..15 -> K=k0+16..23
//   half==1: elems 0..7 -> K=k0+8..15, elems 8..15 -> K=k0+24..31
__device__ __forceinline__ v16bf load_frag(const unsigned short* row, int k0, int half) {
    union { v16bf v; v4u q[2]; } f;
    f.q[0] = *(const v4u*)(row + k0 + half * 8);
    f.q[1] = *(const v4u*)(row + k0 + 16 + half * 8);
    return f.v;
}

// ---------------- utility kernels ----------------

__global__ void cvt_f32_bf16_kernel(const float* __restrict__ in,
                                    unsigned short* __restrict__ out, int n) {
    int i = blockIdx.x * blockDim.x + threadIdx.x;
    if (i < n) out[i] = f2bf(in[i]);
}

__global__ void zero_state_kernel(float* __restrict__ h,
                                  unsigned short* __restrict__ hb, int n) {
    int i = blockIdx.x * blockDim.x + threadIdx.x;
    if (i < n) { h[i] = 0.0f; hb[i] = 0; }
}

// Embedding: emb[row, e] = thresh(W1[e, ids[row]] + b1[e]) -> bf16
__global__ void embed_kernel(const int* __restrict__ ids,
                             const float* __restrict__ W1,
                             const float* __restrict__ b1,
                             unsigned short* __restrict__ emb) {
    int i = blockIdx.x * blockDim.x + threadIdx.x;   // SB_*EH_ threads
    int row = i >> 9;            // / EH_
    int e   = i & (EH_ - 1);
    int id  = ids[row];
    float v = W1[(size_t)e * V_ + id] + b1[e];
    v = (v > 1e-6f) ? v : 0.0f;
    emb[i] = f2bf(v);
}

// ---------------- common WMMA epilogue ----------------
// acc[0..3]: rows m0..m0+15, acc[4..7]: rows m0+16..m0+31 (4 N tiles each).
// C/D layout: lane -> N=(lane&15)+tile, VGPR r -> M = r + 8*half.
template <bool THRESH, bool WF32, bool WBF16>
__device__ __forceinline__ void store_tile(const v8f* acc, const float* __restrict__ bias,
                                           float* __restrict__ Cf, unsigned short* __restrict__ Cb,
                                           int m0, int n0, int N, int lr, int half) {
#pragma unroll
    for (int t = 0; t < 4; ++t) {
        int   n  = n0 + t * 16 + lr;
        float bv = bias[n];
#pragma unroll
        for (int r = 0; r < 8; ++r) {
#pragma unroll
            for (int g = 0; g < 2; ++g) {
                int   m = m0 + g * 16 + r + 8 * half;
                float v = acc[g * 4 + t][r] + bv;
                if (THRESH) v = (v > 1e-6f) ? v : 0.0f;
                size_t off = (size_t)m * N + n;
                if (WF32)  Cf[off] = v;
                if (WBF16) Cb[off] = f2bf(v);
            }
        }
    }
}

// ---------------- WMMA GEMM (direct-from-global, big M) ----------------
// C[m,n] = act( sum_k A[m,k]*W[n,k] + bias[n] );  A:[M,K] bf16, W:[N,K] bf16.
// One wave per block, 32x64 output tile: 2 A frags x 4 B frags -> 8 wmma / K-step.
// Requires M%32==0, N%64==0, K%32==0; EXEC all-ones (blockDim.x==32, no divergence).
template <bool THRESH, bool WF32, bool WBF16>
__global__ void wmma_gemm_kernel(const unsigned short* __restrict__ A,
                                 const unsigned short* __restrict__ W,
                                 const float* __restrict__ bias,
                                 float* __restrict__ Cf,
                                 unsigned short* __restrict__ Cb,
                                 int M, int N, int K) {
    const int lane = threadIdx.x;
    const int half = lane >> 4;
    const int lr   = lane & 15;
    const int m0   = blockIdx.y * 32;
    const int n0   = blockIdx.x * 64;

    const unsigned short* arow0 = A + (size_t)(m0 + lr) * K;
    const unsigned short* arow1 = A + (size_t)(m0 + 16 + lr) * K;
    const unsigned short* wrow[4];
#pragma unroll
    for (int t = 0; t < 4; ++t)
        wrow[t] = W + (size_t)(n0 + t * 16 + lr) * K;

    v8f acc[8];
#pragma unroll
    for (int i = 0; i < 8; ++i) acc[i] = v8f{};

    for (int k0 = 0; k0 < K; k0 += 32) {
        v16bf a0 = load_frag(arow0, k0, half);
        v16bf a1 = load_frag(arow1, k0, half);
#pragma unroll
        for (int t = 0; t < 4; ++t) {
            v16bf b = load_frag(wrow[t], k0, half);
            acc[t]     = __builtin_amdgcn_wmma_f32_16x16x32_bf16(false, a0, false, b,
                                                                 (short)0, acc[t], false, false);
            acc[4 + t] = __builtin_amdgcn_wmma_f32_16x16x32_bf16(false, a1, false, b,
                                                                 (short)0, acc[4 + t], false, false);
        }
    }
    store_tile<THRESH, WF32, WBF16>(acc, bias, Cf, Cb, m0, n0, N, lr, half);
}

// ---------------- WMMA GEMM with async-LDS weight staging (small M) ----------------
// Same math; the 64x32 weight tile of each K-step is double-buffered in LDS via
// global_load_async_to_lds_b128 (ASYNCcnt), B fragments then come from ds_load_b128.
// LDS row stride is 40 elems (80B) so fragment reads hit distinct bank groups.
#define WT_STRIDE 40                     // elems per LDS tile row (64B data + 16B pad)
#define WT_ELEMS  (64 * WT_STRIDE)       // one buffer: 2560 elems = 5120 B

__device__ __forceinline__ void stage_wtile(const unsigned short* __restrict__ W,
                                            int K, int n0, int k0,
                                            unsigned lds_base, int lane) {
#pragma unroll
    for (int c = 0; c < 8; ++c) {        // 8 async b128 ops: 32 lanes * 16B * 8 = 4KB tile
        int id  = c * 32 + lane;         // 256 16B-chunks
        int row = id >> 2;               // 0..63  (n within tile)
        int col = id & 3;                // 0..3   (16B chunk within 64B row)
        unsigned long long g =
            (unsigned long long)(uintptr_t)(W + (size_t)(n0 + row) * K + k0 + col * 8);
        unsigned l = lds_base + (unsigned)(row * (WT_STRIDE * 2) + col * 16);
        asm volatile("global_load_async_to_lds_b128 %0, %1, off"
                     :: "v"(l), "v"(g) : "memory");
    }
}

template <bool THRESH, bool WF32, bool WBF16>
__global__ void wmma_gemm_lds_kernel(const unsigned short* __restrict__ A,
                                     const unsigned short* __restrict__ W,
                                     const float* __restrict__ bias,
                                     float* __restrict__ Cf,
                                     unsigned short* __restrict__ Cb,
                                     int M, int N, int K) {
    __shared__ unsigned short wtile[2][WT_ELEMS];

    const int lane = threadIdx.x;
    const int half = lane >> 4;
    const int lr   = lane & 15;
    const int m0   = blockIdx.y * 32;
    const int n0   = blockIdx.x * 64;

    const unsigned short* arow0 = A + (size_t)(m0 + lr) * K;
    const unsigned short* arow1 = A + (size_t)(m0 + 16 + lr) * K;
    const unsigned lds0 = (unsigned)(uintptr_t)(&wtile[0][0]);
    const unsigned lds1 = (unsigned)(uintptr_t)(&wtile[1][0]);

    v8f acc[8];
#pragma unroll
    for (int i = 0; i < 8; ++i) acc[i] = v8f{};

    stage_wtile(W, K, n0, 0, lds0, lane);

    for (int k0 = 0; k0 < K; k0 += 32) {
        const int buf = (k0 >> 5) & 1;
        if (k0 + 32 < K) {
            stage_wtile(W, K, n0, k0 + 32, buf ? lds0 : lds1, lane);
            asm volatile("s_wait_asynccnt 0x8" ::: "memory");   // current buffer complete
        } else {
            asm volatile("s_wait_asynccnt 0x0" ::: "memory");   // drain (last tile)
        }
        v16bf a0 = load_frag(arow0, k0, half);
        v16bf a1 = load_frag(arow1, k0, half);
        const unsigned short* wt = &wtile[buf][0];
#pragma unroll
        for (int t = 0; t < 4; ++t) {
            // B fragment from LDS: row (t*16+lr) of the tile -> ds_load_b128 x2
            v16bf b = load_frag(wt + (t * 16 + lr) * WT_STRIDE, 0, half);
            acc[t]     = __builtin_amdgcn_wmma_f32_16x16x32_bf16(false, a0, false, b,
                                                                 (short)0, acc[t], false, false);
            acc[4 + t] = __builtin_amdgcn_wmma_f32_16x16x32_bf16(false, a1, false, b,
                                                                 (short)0, acc[4 + t], false, false);
        }
    }
    store_tile<THRESH, WF32, WBF16>(acc, bias, Cf, Cb, m0, n0, N, lr, half);
}

// ---------------- GRU gate kernel (one step) ----------------
__global__ void gru_gate_kernel(const float* __restrict__ xw,
                                const float* __restrict__ gh,
                                float* __restrict__ h,
                                unsigned short* __restrict__ hb) {
    int i = blockIdx.x * blockDim.x + threadIdx.x;   // B_*H_ threads
    int b = i >> 10;             // / H_
    int j = i & (H_ - 1);
    const float* xr_ = xw + (size_t)b * G3H_;
    const float* hr_ = gh + (size_t)b * G3H_;
    float xr = xr_[j], xz = xr_[H_ + j], xn = xr_[2 * H_ + j];
    float hr = hr_[j], hz = hr_[H_ + j], hn = hr_[2 * H_ + j];
    float r = 1.0f / (1.0f + __expf(-(xr + hr)));
    float z = 1.0f / (1.0f + __expf(-(xz + hz)));
    float n = tanhf(xn + r * hn);
    float hv   = h[i];
    float hnew = (1.0f - z) * n + z * hv;
    h[i]  = hnew;
    hb[i] = f2bf(hnew);
}

// ---------------- launch ----------------

extern "C" void kernel_launch(void* const* d_in, const int* in_sizes, int n_in,
                              void* d_out, int out_size, void* d_ws, size_t ws_size,
                              hipStream_t stream) {
    (void)in_sizes; (void)n_in; (void)out_size; (void)ws_size;

    const int*   ids   = (const int*)  d_in[0];
    const float* W1    = (const float*)d_in[1];
    const float* b1    = (const float*)d_in[2];
    const float* W2    = (const float*)d_in[3];
    const float* b2    = (const float*)d_in[4];
    const float* W_ih  = (const float*)d_in[5];
    const float* b_ih  = (const float*)d_in[6];
    const float* W_hh  = (const float*)d_in[7];
    const float* b_hh  = (const float*)d_in[8];
    const float* W_out = (const float*)d_in[9];
    const float* b_out = (const float*)d_in[10];
    float* out = (float*)d_out;

    // ---- workspace layout ----
    char* w = (char*)d_ws;
    size_t off = 0;
    unsigned short* W2b   = (unsigned short*)(w + off); off += (size_t)EO_  * EH_ * 2;
    unsigned short* Wihb  = (unsigned short*)(w + off); off += (size_t)G3H_ * EO_ * 2;
    unsigned short* Whhb  = (unsigned short*)(w + off); off += (size_t)G3H_ * H_  * 2;
    unsigned short* Woutb = (unsigned short*)(w + off); off += (size_t)O_   * H_  * 2;
    unsigned short* embb  = (unsigned short*)(w + off); off += (size_t)SB_  * EH_ * 2;
    unsigned short* xb    = (unsigned short*)(w + off); off += (size_t)SB_  * EO_ * 2;
    float*          xW    = (float*)         (w + off); off += (size_t)SB_  * G3H_* 4;
    float*          gh    = (float*)         (w + off); off += (size_t)B_   * G3H_* 4;
    float*          h     = (float*)         (w + off); off += (size_t)B_   * H_  * 4;
    unsigned short* hb    = (unsigned short*)(w + off); off += (size_t)B_   * H_  * 2;

    // 1) weights -> bf16
    {
        int n;
        n = EO_ * EH_;   cvt_f32_bf16_kernel<<<(n + 255) / 256, 256, 0, stream>>>(W2,    W2b,   n);
        n = G3H_ * EO_;  cvt_f32_bf16_kernel<<<(n + 255) / 256, 256, 0, stream>>>(W_ih,  Wihb,  n);
        n = G3H_ * H_;   cvt_f32_bf16_kernel<<<(n + 255) / 256, 256, 0, stream>>>(W_hh,  Whhb,  n);
        n = O_ * H_;     cvt_f32_bf16_kernel<<<(n + 255) / 256, 256, 0, stream>>>(W_out, Woutb, n);
    }

    // 2) embedding gather + bias + threshold -> bf16 [SB, EH]
    embed_kernel<<<(SB_ * EH_) / 256, 256, 0, stream>>>(ids, W1, b1, embb);

    // 3) x = thresh(emb @ W2^T + b2) -> bf16 [SB, EO]
    wmma_gemm_kernel<true, false, true>
        <<<dim3(EO_ / 64, SB_ / 32), 32, 0, stream>>>(embb, W2b, b2, nullptr, xb,
                                                      SB_, EO_, EH_);

    // 4) xW = x @ W_ih^T + b_ih -> f32 [SB, 3H]
    wmma_gemm_kernel<false, true, false>
        <<<dim3(G3H_ / 64, SB_ / 32), 32, 0, stream>>>(xb, Wihb, b_ih, xW, nullptr,
                                                       SB_, G3H_, EO_);

    // 5) h = 0
    zero_state_kernel<<<(B_ * H_ + 255) / 256, 256, 0, stream>>>(h, hb, B_ * H_);

    // 6) sequential GRU: 128 steps of {gh = h @ W_hh^T + b_hh ; gates}
    //    step GEMM uses the async-LDS weight-staging kernel (weights are L2-hot).
    for (int s = 0; s < S_; ++s) {
        wmma_gemm_lds_kernel<false, true, false>
            <<<dim3(G3H_ / 64, B_ / 32), 32, 0, stream>>>(hb, Whhb, b_hh, gh, nullptr,
                                                          B_, G3H_, H_);
        gru_gate_kernel<<<(B_ * H_) / 256, 256, 0, stream>>>(
            xW + (size_t)s * B_ * G3H_, gh, h, hb);
    }

    // 7) out = hT @ W_out^T + b_out -> f32 [1, B, O]
    wmma_gemm_lds_kernel<false, true, false>
        <<<dim3(O_ / 64, B_ / 32), 32, 0, stream>>>(hb, Woutb, b_out, out, nullptr,
                                                    B_, O_, H_);
}
